// Transformer_46059229283095
// MI455X (gfx1250) — matrix operand order repaired and checked by reading, not compiled, not run on hardware
//
#include <hip/hip_runtime.h>
#include <stdint.h>

// ---------------------------------------------------------------------------
// Transformer encoder forward for gfx1250 (MI455X), bf16 WMMA + f32 accumulate
// N=8, L=1024, D=512, H=8 (E=64), FF=2048, 4 layers.
// GEMM: 128x128x32 block tile, 8 waves x (32x64), double-buffered LDS with
// GLOBAL_LOAD_ASYNC_TO_LDS_B128 staging (ASYNCcnt pipeline) when available.
// ---------------------------------------------------------------------------

typedef __bf16 v16bf __attribute__((ext_vector_type(16)));
typedef __bf16 v8bf  __attribute__((ext_vector_type(8)));
typedef float  v8f   __attribute__((ext_vector_type(8)));
typedef int    v4i   __attribute__((ext_vector_type(4)));
typedef unsigned short u16;

typedef __attribute__((address_space(1))) v4i gv4i;   // global
typedef __attribute__((address_space(3))) v4i lv4i;   // LDS

#define N_BATCH 8
#define SEQ_L   1024
#define DMODEL  512
#define NHEADS  8
#define EHEAD   64
#define DFF     2048
#define NLAYERS 4
#define NTOK    (N_BATCH * SEQ_L)   // 8192

// ---------------- CDNA5 async global->LDS copy (ASYNCcnt) -------------------
#if __has_builtin(__builtin_amdgcn_global_load_async_to_lds_b128)
#define HAVE_ASYNC_LDS 1
static __device__ __forceinline__ void async_b128(const u16* g, u16* l) {
    // AS1: flat == global address. AS3: LDS offset == flat_addr[31:0] (ISA 10.2).
    __builtin_amdgcn_global_load_async_to_lds_b128(
        (gv4i*)(unsigned long long)(uintptr_t)g,
        (lv4i*)(unsigned int)(uintptr_t)l, 0, 0);
}
template <int N> static __device__ __forceinline__ void wait_async() {
#if __has_builtin(__builtin_amdgcn_s_wait_asynccnt)
    __builtin_amdgcn_s_wait_asynccnt(N);
#else
    asm volatile("s_wait_asynccnt %0" ::"n"(N) : "memory");
#endif
}
#else
#define HAVE_ASYNC_LDS 0
#endif

// float -> bf16, round-to-nearest-even
static __device__ __forceinline__ u16 f2bf(float f) {
    unsigned u = __float_as_uint(f);
    u += 0x7FFFu + ((u >> 16) & 1u);
    return (u16)(u >> 16);
}

static __device__ __forceinline__ v16bf frag_cat(const u16* p0, const u16* p1) {
    v8bf a = *(const v8bf*)p0;
    v8bf b = *(const v8bf*)p1;
    v16bf r;
#pragma unroll
    for (int i = 0; i < 8; ++i) { r[i] = a[i]; r[i + 8] = b[i]; }
    return r;
}

// A-operand fragment (16x32 window): tile is row-major [rows][stride] bf16.
// ISA: lanes 0-15 -> M=lane, K {0..7,16..23}; lanes 16-31 -> K {8..15,24..31}.
static __device__ __forceinline__ v16bf fragA(const u16* base, int stride,
                                              int mo, int kbase, int lane) {
    const u16* p = base + (mo + (lane & 15)) * stride + kbase + (lane >> 4) * 8;
    return frag_cat(p, p + 16);
}

// B-operand fragment (32x16 window): tile is N-major [cols][stride] bf16.
// ISA: lanes 0-15 -> N=lane, K {0..7,16..23}; lanes 16-31 -> N=lane-16, K {8..15,24..31}.
static __device__ __forceinline__ v16bf fragB(const u16* base, int stride,
                                              int no, int kbase, int lane) {
    const u16* p = base + (no + (lane & 15)) * stride + kbase + (lane >> 4) * 8;
    return frag_cat(p, p + 16);
}

static __device__ __forceinline__ v8f wmma_bf16(v16bf a, v16bf b, v8f c) {
    return __builtin_amdgcn_wmma_f32_16x16x32_bf16(false, a, false, b,
                                                   (short)0, c, false, false);
}

// ---------------------------------------------------------------------------
// Weight transpose + convert: dst[n*K + k] = bf16(src[k*N + n])   (N-major)
// ---------------------------------------------------------------------------
__global__ __launch_bounds__(256) void tconv_kernel(u16* __restrict__ dst,
                                                    const float* __restrict__ src,
                                                    int K, int N) {
    int idx = blockIdx.x * 256 + threadIdx.x;
    if (idx >= K * N) return;
    int k = idx / N, n = idx - k * N;
    dst[(size_t)n * K + k] = f2bf(src[idx]);
}

__global__ __launch_bounds__(256) void copyx_kernel(float* __restrict__ hf,
                                                    u16* __restrict__ hb,
                                                    const float* __restrict__ x,
                                                    int n) {
    int i = blockIdx.x * 256 + threadIdx.x;
    if (i >= n) return;
    float v = x[i];
    hf[i] = v;
    hb[i] = f2bf(v);
}

// ---------------------------------------------------------------------------
// Tiled GEMM:  C[M,N] = A[M,K](bf16,row-major) @ Bt[N,K](bf16,N-major) + bias
// Block 128x128x32, 8 waves each 32x64 (2x4 wmma accs), double-buffered LDS.
// ---------------------------------------------------------------------------
#define BM 128
#define BN 128
#define BK 32
#define TSTR 40                    // LDS row stride (bf16 units), 80B = 16B-mult
#define TILE_N (128 * TSTR)

template <int OUTBF16, int RELU, int RESID>
__global__ __launch_bounds__(256) void gemm_wmma(const u16* __restrict__ A,
                                                 const u16* __restrict__ Bt,
                                                 const float* __restrict__ bias,
                                                 const float* __restrict__ resid,
                                                 float* __restrict__ Cf,
                                                 u16* __restrict__ Cb,
                                                 int M, int N, int K) {
    __shared__ u16 As[2][TILE_N];
    __shared__ u16 Bs[2][TILE_N];

    const int tid  = threadIdx.x;
    const int lane = tid & 31;
    const int w    = tid >> 5;                 // 0..7
    const int m0   = blockIdx.y * BM;
    const int n0   = blockIdx.x * BN;
    const int wm   = (w & 3) * 32;             // wave row offset
    const int wn   = (w >> 2) * 64;            // wave col offset

    v8f acc[2][4];
#pragma unroll
    for (int i = 0; i < 2; ++i)
#pragma unroll
        for (int j = 0; j < 4; ++j)
#pragma unroll
            for (int r = 0; r < 8; ++r) acc[i][j][r] = 0.0f;

    // staging: 128 rows x 32 bf16 per tile; 2 threads/row, 2x b128 each
    const int srow = tid >> 1;                 // 0..127
    const int scol = (tid & 1) * 16;           // 0,16 (bf16 units)
    const u16* ga = A  + (size_t)(m0 + srow) * K + scol;
    const u16* gb = Bt + (size_t)(n0 + srow) * K + scol;
    const int   so = srow * TSTR + scol;
    const int   NK = K / BK;

#if HAVE_ASYNC_LDS
    // prologue: stream tile 0 into buffer 0 via async DMA
    async_b128(ga, &As[0][so]);
    async_b128(ga + 8, &As[0][so + 8]);
    async_b128(gb, &Bs[0][so]);
    async_b128(gb + 8, &Bs[0][so + 8]);
    for (int kt = 0; kt < NK; ++kt) {
        const int cur = kt & 1;
        if (kt + 1 < NK) {            // stream next tile into other buffer
            const int ko = (kt + 1) * BK;
            async_b128(ga + ko, &As[1 - cur][so]);
            async_b128(ga + ko + 8, &As[1 - cur][so + 8]);
            async_b128(gb + ko, &Bs[1 - cur][so]);
            async_b128(gb + ko + 8, &Bs[1 - cur][so + 8]);
            wait_async<4>();          // current tile (issued first) complete
        } else {
            wait_async<0>();
        }
        __syncthreads();
#else
    int4 ra0, ra1, rb0, rb1;
    ra0 = *(const int4*)ga;       ra1 = *(const int4*)(ga + 8);
    rb0 = *(const int4*)gb;       rb1 = *(const int4*)(gb + 8);
    for (int kt = 0; kt < NK; ++kt) {
        const int cur = kt & 1;
        *(int4*)&As[cur][so] = ra0;  *(int4*)&As[cur][so + 8] = ra1;
        *(int4*)&Bs[cur][so] = rb0;  *(int4*)&Bs[cur][so + 8] = rb1;
        if (kt + 1 < NK) {            // issue next tile's global loads now
            const int ko = (kt + 1) * BK;
            ra0 = *(const int4*)(ga + ko);  ra1 = *(const int4*)(ga + ko + 8);
            rb0 = *(const int4*)(gb + ko);  rb1 = *(const int4*)(gb + ko + 8);
        }
        __syncthreads();
#endif
        // 8 wmma per wave per K-step
        v16bf bfr[4];
#pragma unroll
        for (int j = 0; j < 4; ++j)
            bfr[j] = fragB(Bs[cur], TSTR, wn + j * 16, 0, lane);
#pragma unroll
        for (int i = 0; i < 2; ++i) {
            v16bf af = fragA(As[cur], TSTR, wm + i * 16, 0, lane);
#pragma unroll
            for (int j = 0; j < 4; ++j)
                acc[i][j] = wmma_bf16(af, bfr[j], acc[i][j]);
        }
        __syncthreads();
    }

    // epilogue: C/D layout lane->col (l&15), vgpr r -> row r + 8*(l>>4)
    const int rofs = 8 * (lane >> 4);
    const int cl   = lane & 15;
#pragma unroll
    for (int i = 0; i < 2; ++i)
#pragma unroll
        for (int j = 0; j < 4; ++j) {
            const int n = n0 + wn + j * 16 + cl;
            const float bv = bias[n];
#pragma unroll
            for (int r = 0; r < 8; ++r) {
                const int m = m0 + wm + i * 16 + r + rofs;
                float v = acc[i][j][r] + bv;
                if (RELU) v = fmaxf(v, 0.0f);
                if (RESID) v += resid[(size_t)m * N + n];
                if (OUTBF16) Cb[(size_t)m * N + n] = f2bf(v);
                else         Cf[(size_t)m * N + n] = v;
            }
        }
}

// ---------------------------------------------------------------------------
// Flash attention: per (batch*head, 64-row Q block), loop 64-wide K/V chunks.
// Q,K,V,O are bf16 [NTOK, DMODEL]; head h uses columns [h*64, h*64+64).
// ---------------------------------------------------------------------------
#define QSTR 72
#define SSTR 66

__global__ __launch_bounds__(256) void flash_attn_kernel(const u16* __restrict__ Qg,
                                                         const u16* __restrict__ Kg,
                                                         const u16* __restrict__ Vg,
                                                         u16* __restrict__ Og) {
    __shared__ u16  Qs[64 * QSTR];
    __shared__ u16  Ks[64 * QSTR];   // reused as P (bf16 probabilities)
    __shared__ u16  Vt[64 * QSTR];   // V transposed: Vt[e][s]
    __shared__ float Ss[64 * SSTR];  // raw scores (f32)
    __shared__ float mrun[64], lrun[64], msc[64];

    const int bh  = blockIdx.x;            // n*H + h
    const int nb  = bh >> 3;
    const int hh  = bh & 7;
    const int q0  = blockIdx.y * 64;
    const int tid = threadIdx.x, lane = tid & 31, w = tid >> 5;
    const float scale = 0.125f;            // 1/sqrt(64)

    const size_t headoff = (size_t)hh * EHEAD;
    const u16* Qb = Qg + (size_t)nb * SEQ_L * DMODEL + headoff;
    const u16* Kb = Kg + (size_t)nb * SEQ_L * DMODEL + headoff;
    const u16* Vb = Vg + (size_t)nb * SEQ_L * DMODEL + headoff;

    const int srow = tid >> 2, sseg = (tid & 3) * 16;
    const int qso  = srow * QSTR + sseg;

    // load Q block: 64 rows x 64 bf16
#if HAVE_ASYNC_LDS
    async_b128(&Qb[(size_t)(q0 + srow) * DMODEL + sseg], &Qs[qso]);
    async_b128(&Qb[(size_t)(q0 + srow) * DMODEL + sseg + 8], &Qs[qso + 8]);
#else
    *(int4*)&Qs[qso]     = *(const int4*)&Qb[(size_t)(q0 + srow) * DMODEL + sseg];
    *(int4*)&Qs[qso + 8] = *(const int4*)&Qb[(size_t)(q0 + srow) * DMODEL + sseg + 8];
#endif
    if (tid < 64) { mrun[tid] = -1e30f; lrun[tid] = 0.0f; }

    const int moff = (w >> 1) * 16;        // wave row offset (S and O)
    const int noff = (w & 1) * 32;         // wave col offset (S and O)
    const int rofs = 8 * (lane >> 4);
    const int cl   = lane & 15;

    v8f o[2];
#pragma unroll
    for (int j = 0; j < 2; ++j)
#pragma unroll
        for (int r = 0; r < 8; ++r) o[j][r] = 0.0f;

    for (int c0 = 0; c0 < SEQ_L; c0 += 64) {
        __syncthreads();   // prior P@V reads of Ks/Vt complete
        // stage K chunk (row-major, async) and V chunk (register transpose)
        {
            const size_t gro = (size_t)(c0 + srow) * DMODEL + sseg;
#if HAVE_ASYNC_LDS
            async_b128(&Kb[gro], &Ks[qso]);
            async_b128(&Kb[gro + 8], &Ks[qso + 8]);
#else
            *(int4*)&Ks[qso]     = *(const int4*)&Kb[gro];
            *(int4*)&Ks[qso + 8] = *(const int4*)&Kb[gro + 8];
#endif
            union { int4 v[2]; u16 u[16]; } tv;
            tv.v[0] = *(const int4*)&Vb[gro];
            tv.v[1] = *(const int4*)&Vb[gro + 8];
#pragma unroll
            for (int e = 0; e < 16; ++e) Vt[(sseg + e) * QSTR + srow] = tv.u[e];
#if HAVE_ASYNC_LDS
            wait_async<0>();   // Q (first iter) + K chunk landed in LDS
#endif
        }
        __syncthreads();

        // S = Q @ K^T  (K-tile row-major == N-major B operand)
        v8f sacc[2];
#pragma unroll
        for (int j = 0; j < 2; ++j)
#pragma unroll
            for (int r = 0; r < 8; ++r) sacc[j][r] = 0.0f;
#pragma unroll
        for (int ks = 0; ks < 64; ks += 32) {
            v16bf aq = fragA(Qs, QSTR, moff, ks, lane);
#pragma unroll
            for (int j = 0; j < 2; ++j) {
                v16bf bk = fragB(Ks, QSTR, noff + j * 16, ks, lane);
                sacc[j] = wmma_bf16(aq, bk, sacc[j]);
            }
        }
#pragma unroll
        for (int j = 0; j < 2; ++j)
#pragma unroll
            for (int r = 0; r < 8; ++r)
                Ss[(moff + r + rofs) * SSTR + noff + j * 16 + cl] = sacc[j][r] * scale;
        __syncthreads();   // Ss ready; Ks free for reuse as P

        // online softmax stats, P = exp(S - m_new) into Ks
        if (tid < 64) {
            const int r = tid;
            float mold = mrun[r], cm = -1e30f;
            for (int jj = 0; jj < 64; ++jj) cm = fmaxf(cm, Ss[r * SSTR + jj]);
            float mn = fmaxf(mold, cm);
            float al = __expf(mold - mn);
            float sum = 0.0f;
            for (int jj = 0; jj < 64; ++jj) {
                float p = __expf(Ss[r * SSTR + jj] - mn);
                sum += p;
                Ks[r * QSTR + jj] = f2bf(p);
            }
            lrun[r] = lrun[r] * al + sum;
            mrun[r] = mn;
            msc[r]  = al;
        }
        __syncthreads();   // P, msc ready

        // rescale O accumulators, then O += P @ V
#pragma unroll
        for (int j = 0; j < 2; ++j)
#pragma unroll
            for (int r = 0; r < 8; ++r) o[j][r] *= msc[moff + r + rofs];
#pragma unroll
        for (int ks = 0; ks < 64; ks += 32) {
            v16bf ap = fragA(Ks, QSTR, moff, ks, lane);
#pragma unroll
            for (int j = 0; j < 2; ++j) {
                v16bf bv = fragB(Vt, QSTR, noff + j * 16, ks, lane);
                o[j] = wmma_bf16(ap, bv, o[j]);
            }
        }
    }

    __syncthreads();
#pragma unroll
    for (int j = 0; j < 2; ++j)
#pragma unroll
        for (int r = 0; r < 8; ++r) {
            const int m = moff + r + rofs;
            const int e = noff + j * 16 + cl;
            float v = o[j][r] / lrun[m];
            Og[(size_t)(nb * SEQ_L + q0 + m) * DMODEL + headoff + e] = f2bf(v);
        }
}

// ---------------------------------------------------------------------------
// LayerNorm: one wave per token row (D=512); writes f32 + bf16 copies.
// ---------------------------------------------------------------------------
__global__ __launch_bounds__(256) void layernorm_kernel(const float* __restrict__ x,
                                                        const float* __restrict__ g,
                                                        const float* __restrict__ b,
                                                        float* __restrict__ yf,
                                                        u16* __restrict__ yb) {
    const int row  = blockIdx.x * 8 + (threadIdx.x >> 5);
    const int lane = threadIdx.x & 31;
    const float4* xr = (const float4*)(x + (size_t)row * DMODEL);
    float4 v[4];
    float s = 0.0f, sq = 0.0f;
#pragma unroll
    for (int j = 0; j < 4; ++j) {
        v[j] = xr[lane + 32 * j];
        s  += v[j].x + v[j].y + v[j].z + v[j].w;
        sq += v[j].x * v[j].x + v[j].y * v[j].y + v[j].z * v[j].z + v[j].w * v[j].w;
    }
#pragma unroll
    for (int off = 16; off > 0; off >>= 1) {
        s  += __shfl_xor(s,  off, 32);
        sq += __shfl_xor(sq, off, 32);
    }
    const float mean = s * (1.0f / DMODEL);
    const float var  = sq * (1.0f / DMODEL) - mean * mean;
    const float rs   = rsqrtf(var + 1e-5f);
#pragma unroll
    for (int j = 0; j < 4; ++j) {
        const int c4 = (lane + 32 * j) * 4;
        float4 gv = ((const float4*)g)[lane + 32 * j];
        float4 bv = ((const float4*)b)[lane + 32 * j];
        float y0 = (v[j].x - mean) * rs * gv.x + bv.x;
        float y1 = (v[j].y - mean) * rs * gv.y + bv.y;
        float y2 = (v[j].z - mean) * rs * gv.z + bv.z;
        float y3 = (v[j].w - mean) * rs * gv.w + bv.w;
        float4 yv; yv.x = y0; yv.y = y1; yv.z = y2; yv.w = y3;
        *(float4*)(yf + (size_t)row * DMODEL + c4) = yv;
        ushort4 yh;
        yh.x = f2bf(y0); yh.y = f2bf(y1); yh.z = f2bf(y2); yh.w = f2bf(y3);
        *(ushort4*)(yb + (size_t)row * DMODEL + c4) = yh;
    }
}

// ---------------------------------------------------------------------------
// Host orchestration
// ---------------------------------------------------------------------------
extern "C" void kernel_launch(void* const* d_in, const int* in_sizes, int n_in,
                              void* d_out, int out_size, void* d_ws, size_t ws_size,
                              hipStream_t stream) {
    const float* x  = (const float*)d_in[0];
    const float* Wq = (const float*)d_in[1];
    const float* bq = (const float*)d_in[2];
    const float* Wk = (const float*)d_in[3];
    const float* bk = (const float*)d_in[4];
    const float* Wv = (const float*)d_in[5];
    const float* bv = (const float*)d_in[6];
    const float* Wo = (const float*)d_in[7];
    const float* bo = (const float*)d_in[8];
    const float* W1 = (const float*)d_in[9];
    const float* b1 = (const float*)d_in[10];
    const float* W2 = (const float*)d_in[11];
    const float* b2 = (const float*)d_in[12];
    const float* g1 = (const float*)d_in[13];
    const float* be1= (const float*)d_in[14];
    const float* g2 = (const float*)d_in[15];
    const float* be2= (const float*)d_in[16];

    size_t off = 0;
    auto alloc = [&](size_t bytes) -> void* {
        void* p = (char*)d_ws + off;
        off = (off + bytes + 255) & ~(size_t)255;
        return p;
    };
    const size_t TD  = (size_t)NTOK * DMODEL;
    const size_t TF  = (size_t)NTOK * DFF;
    float* hbuf  = (float*)alloc(TD * 4);     // hidden state (f32)
    float* hln   = (float*)alloc(TD * 4);     // post-LN1 (f32, residual for FFN2)
    u16* hbf     = (u16*)alloc(TD * 2);       // hidden state (bf16)
    u16* hlnbf   = (u16*)alloc(TD * 2);       // post-LN1 (bf16)
    u16* qb      = (u16*)alloc(TD * 2);
    u16* kb      = (u16*)alloc(TD * 2);
    u16* vb      = (u16*)alloc(TD * 2);
    u16* attnb   = (u16*)alloc(TD * 2);
    u16* ffn1b   = (u16*)alloc(TF * 2);
    const size_t WDD = (size_t)DMODEL * DMODEL;
    const size_t WDF = (size_t)DMODEL * DFF;
    u16* wqt = (u16*)alloc(NLAYERS * WDD * 2);
    u16* wkt = (u16*)alloc(NLAYERS * WDD * 2);
    u16* wvt = (u16*)alloc(NLAYERS * WDD * 2);
    u16* wot = (u16*)alloc(NLAYERS * WDD * 2);
    u16* w1t = (u16*)alloc(NLAYERS * WDF * 2);
    u16* w2t = (u16*)alloc(NLAYERS * WDF * 2);

    const int gdd = (int)(WDD / 256), gdf = (int)(WDF / 256);
    for (int l = 0; l < NLAYERS; ++l) {
        tconv_kernel<<<gdd, 256, 0, stream>>>(wqt + l * WDD, Wq + l * WDD, DMODEL, DMODEL);
        tconv_kernel<<<gdd, 256, 0, stream>>>(wkt + l * WDD, Wk + l * WDD, DMODEL, DMODEL);
        tconv_kernel<<<gdd, 256, 0, stream>>>(wvt + l * WDD, Wv + l * WDD, DMODEL, DMODEL);
        tconv_kernel<<<gdd, 256, 0, stream>>>(wot + l * WDD, Wo + l * WDD, DMODEL, DMODEL);
        tconv_kernel<<<gdf, 256, 0, stream>>>(w1t + l * WDF, W1 + l * WDF, DMODEL, DFF);
        tconv_kernel<<<gdf, 256, 0, stream>>>(w2t + l * WDF, W2 + l * WDF, DFF, DMODEL);
    }

    copyx_kernel<<<(int)(TD / 256), 256, 0, stream>>>(hbuf, hbf, x, (int)TD);

    const dim3 gD(DMODEL / BN, NTOK / BM);           // (4, 64)
    const dim3 gF(DFF / BN, NTOK / BM);              // (16, 64)
    const dim3 gAtt(N_BATCH * NHEADS, SEQ_L / 64);   // (64, 16)

    for (int l = 0; l < NLAYERS; ++l) {
        gemm_wmma<1, 0, 0><<<gD, 256, 0, stream>>>(hbf, wqt + l * WDD, bq + l * DMODEL,
                                                   nullptr, nullptr, qb, NTOK, DMODEL, DMODEL);
        gemm_wmma<1, 0, 0><<<gD, 256, 0, stream>>>(hbf, wkt + l * WDD, bk + l * DMODEL,
                                                   nullptr, nullptr, kb, NTOK, DMODEL, DMODEL);
        gemm_wmma<1, 0, 0><<<gD, 256, 0, stream>>>(hbf, wvt + l * WDD, bv + l * DMODEL,
                                                   nullptr, nullptr, vb, NTOK, DMODEL, DMODEL);
        flash_attn_kernel<<<gAtt, 256, 0, stream>>>(qb, kb, vb, attnb);
        gemm_wmma<0, 0, 1><<<gD, 256, 0, stream>>>(attnb, wot + l * WDD, bo + l * DMODEL,
                                                   hbuf, hbuf, nullptr, NTOK, DMODEL, DMODEL);
        layernorm_kernel<<<NTOK / 8, 256, 0, stream>>>(hbuf, g1 + l * DMODEL, be1 + l * DMODEL,
                                                       hln, hlnbf);
        gemm_wmma<1, 1, 0><<<gF, 256, 0, stream>>>(hlnbf, w1t + l * WDF, b1 + l * DFF,
                                                   nullptr, nullptr, ffn1b, NTOK, DFF, DMODEL);
        gemm_wmma<0, 0, 1><<<gD, 256, 0, stream>>>(ffn1b, w2t + l * WDF, b2 + l * DMODEL,
                                                   hln, hbuf, nullptr, NTOK, DMODEL, DFF);
        layernorm_kernel<<<NTOK / 8, 256, 0, stream>>>(hbuf, g2 + l * DMODEL, be2 + l * DMODEL,
                                                       hbuf, hbf);
    }

    (void)hipMemcpyAsync(d_out, hbuf, TD * sizeof(float), hipMemcpyDeviceToDevice, stream);
}
